// GraphMamba_53652731462317
// MI455X (gfx1250) — matrix operand order repaired and checked by reading, not compiled, not run on hardware
//
#include <hip/hip_runtime.h>
#include <hip/hip_bf16.h>

#define T_DIM 32
#define N_DIM 2048
#define POS_DIM 256
#define H_DIM 64
#define E_DIM 128
#define S_DIM 16
#define R_DIM 4

typedef __attribute__((ext_vector_type(16))) __bf16 v16bf;
typedef __attribute__((ext_vector_type(8)))  __bf16 v8bf;
typedef __attribute__((ext_vector_type(8)))  float  v8f;

__device__ __forceinline__ unsigned short f2bf(float f) {
    unsigned int u = __float_as_uint(f);
    u += 0x7fffu + ((u >> 16) & 1u);   // round-to-nearest-even
    return (unsigned short)(u >> 16);
}

// ---------------------------------------------------------------------------
// msgs = pos_emb @ g1_msg_w + b, stored transposed bf16 [H][N]
// ---------------------------------------------------------------------------
__global__ __launch_bounds__(64)
void prep_msgs_kernel(const float* __restrict__ pos_emb,
                      const float* __restrict__ w, const float* __restrict__ b,
                      unsigned short* __restrict__ msgsT) {
    __shared__ float pe[POS_DIM];
    const int n = blockIdx.x, h = threadIdx.x;
    #pragma unroll
    for (int j = 0; j < POS_DIM / 64; j++)
        pe[h + j * 64] = pos_emb[(size_t)n * POS_DIM + h + j * 64];
    __syncthreads();
    float acc = b[h];
    for (int k = 0; k < POS_DIM; k++) acc += pe[k] * w[k * H_DIM + h];
    msgsT[(size_t)h * N_DIM + n] = f2bf(acc);
}

// ---------------------------------------------------------------------------
// agg[t] = G[t] (fp32 -> bf16) @ B[t]  using v_wmma_f32_16x16x32_bf16
// B is bf16, transposed [H][N] (per-t stride bStrideT; 0 => shared across t)
// Block: 256 threads (8 waves), computes a 128x64 output tile.
// ---------------------------------------------------------------------------
__global__ __launch_bounds__(256)
void gemm_graph_wmma(const float* __restrict__ G,
                     const unsigned short* __restrict__ BT,
                     long long bStrideT,
                     float* __restrict__ out) {
    __shared__ __align__(16) unsigned short lA[128 * 32]; // 8 KB
    __shared__ __align__(16) unsigned short lB[64 * 32];  // 4 KB

    const int t    = blockIdx.y;
    const int row0 = blockIdx.x * 128;
    const int tid  = threadIdx.x;
    const int lane = tid & 31;
    const int wave = tid >> 5;

    const float* Gt = G + (size_t)t * N_DIM * N_DIM;
    const unsigned short* Bt = BT + (size_t)t * bStrideT;

    v8f acc[4];
    #pragma unroll
    for (int nb = 0; nb < 4; nb++) acc[nb] = (v8f){0, 0, 0, 0, 0, 0, 0, 0};

    const int arow  = tid >> 1;   // 0..127: A row within tile
    const int ahalf = tid & 1;    // which 16-float half of the K=32 chunk

    for (int k0 = 0; k0 < N_DIM; k0 += 32) {
        // --- A tile: 128x32 fp32 -> bf16 (convert in registers) ---
        const float4* src =
            (const float4*)(Gt + (size_t)(row0 + arow) * N_DIM + k0 + ahalf * 16);
        float4 f[4];
        #pragma unroll
        for (int i = 0; i < 4; i++) f[i] = src[i];
        unsigned short* dst = &lA[arow * 32 + ahalf * 16];
        const float* fs = (const float*)f;
        #pragma unroll
        for (int i = 0; i < 16; i++) dst[i] = f2bf(fs[i]);

        // --- B tile: 64 cols x 32 K, K-contiguous per column ---
        {
            const int c  = tid >> 2;          // 0..63
            const int ko = (tid & 3) * 8;     // 8 bf16 = 16 bytes
            *(uint4*)&lB[c * 32 + ko] =
                *(const uint4*)(Bt + (size_t)c * N_DIM + k0 + ko);
        }
        __syncthreads();

        // --- fragments: contiguous K order k = 16*(lane/16) + s, same for A and B (wave32) ---
        const int mrow = wave * 16 + (lane & 15);
        const int aoff = mrow * 32 + ((lane < 16) ? 0 : 16);
        v16bf afrag = *(const v16bf*)&lA[aoff];    // K {0..15} or {16..31}

        #pragma unroll
        for (int nb = 0; nb < 4; nb++) {
            const int col = nb * 16 + (lane & 15);
            v16bf bfrag = *(const v16bf*)&lB[col * 32 + ((lane < 16) ? 0 : 16)];
            acc[nb] = __builtin_amdgcn_wmma_f32_16x16x32_bf16(
                false, afrag, false, bfrag, (short)0, acc[nb], false, false);
        }
        __syncthreads();
    }

    // --- store: C/D layout VGPR r: lanes0-15 M=r, lanes16-31 M=r+8, N=lane%16
    const int mbase = row0 + wave * 16 + ((lane < 16) ? 0 : 8);
    const int col16 = lane & 15;
    #pragma unroll
    for (int nb = 0; nb < 4; nb++) {
        #pragma unroll
        for (int r = 0; r < 8; r++) {
            out[(size_t)t * N_DIM * H_DIM + (size_t)(mbase + r) * H_DIM +
                nb * 16 + col16] = acc[nb][r];
        }
    }
}

// ---------------------------------------------------------------------------
// x = LN(relu(concat(rowA, rowB) @ W + bias)); one block per (t,n) row.
// ---------------------------------------------------------------------------
__global__ __launch_bounds__(64)
void upd_ln_kernel(const float* __restrict__ inA, long long tStrideA, int lenA,
                   const float* __restrict__ inB, int lenB,
                   const float* __restrict__ W, const float* __restrict__ bias,
                   const float* __restrict__ lng, const float* __restrict__ lnb,
                   float* __restrict__ out) {
    __shared__ float lin[POS_DIM + H_DIM];
    __shared__ float sh[H_DIM];
    const int n = blockIdx.x, t = blockIdx.y, h = threadIdx.x;
    const float* rowA = inA + (size_t)t * tStrideA + (size_t)n * lenA;
    const float* rowB = inB + ((size_t)t * N_DIM + n) * lenB;
    for (int k = h; k < lenA; k += 64) lin[k] = rowA[k];
    for (int k = h; k < lenB; k += 64) lin[lenA + k] = rowB[k];
    __syncthreads();
    const int K = lenA + lenB;
    float acc = bias[h];
    for (int k = 0; k < K; k++) acc += lin[k] * W[k * H_DIM + h];
    float v = acc > 0.f ? acc : 0.f;
    sh[h] = v;
    __syncthreads();
    float s = 0.f, s2 = 0.f;
    for (int k = 0; k < H_DIM; k++) { float x = sh[k]; s += x; s2 += x * x; }
    const float m   = s * (1.f / H_DIM);
    const float var = s2 * (1.f / H_DIM) - m * m;
    const float r   = rsqrtf(var + 1e-5f);
    out[((size_t)t * N_DIM + n) * H_DIM + h] = (v - m) * r * lng[h] + lnb[h];
}

// ---------------------------------------------------------------------------
// msgs2T[t][h][n] = bf16(x1[t][n] @ g2_msg_w + b)
// ---------------------------------------------------------------------------
__global__ __launch_bounds__(64)
void msgs2_kernel(const float* __restrict__ x1,
                  const float* __restrict__ w, const float* __restrict__ b,
                  unsigned short* __restrict__ m2T) {
    __shared__ float lin[H_DIM];
    const int n = blockIdx.x, t = blockIdx.y, h = threadIdx.x;
    lin[h] = x1[((size_t)t * N_DIM + n) * H_DIM + h];
    __syncthreads();
    float acc = b[h];
    for (int k = 0; k < H_DIM; k++) acc += lin[k] * w[k * H_DIM + h];
    m2T[(size_t)t * H_DIM * N_DIM + (size_t)h * N_DIM + n] = f2bf(acc);
}

// ---------------------------------------------------------------------------
// Mamba over time per node. One block (256 thr) per node; xres = [T][N][H],
// read as input/residual and overwritten in place with the final output.
// ---------------------------------------------------------------------------
__global__ __launch_bounds__(256)
void mamba_kernel(float* xres,
                  const float* __restrict__ mg, const float* __restrict__ mb,
                  const float* __restrict__ in_w, const float* __restrict__ in_b,
                  const float* __restrict__ dp_w, const float* __restrict__ dp_b,
                  const float* __restrict__ dt_w, const float* __restrict__ dt_b,
                  const float* __restrict__ A_log, const float* __restrict__ Dpv,
                  const float* __restrict__ B_w, const float* __restrict__ B_b,
                  const float* __restrict__ C_w, const float* __restrict__ C_b,
                  const float* __restrict__ out_w, const float* __restrict__ out_b) {
    const int n = blockIdx.x;
    const int tid = threadIdx.x;
    __shared__ float xrow[H_DIM];
    __shared__ float xn[H_DIM];
    __shared__ float xp[E_DIM];
    __shared__ float gs[E_DIM];
    __shared__ float del[E_DIM];
    __shared__ float dpr[R_DIM];
    __shared__ float Bm[S_DIM], Cm[S_DIM];
    __shared__ float ypart[256];
    __shared__ float yv[E_DIM];

    // per-thread persistent: 8 SSM state cells (e = tid/2, s half by tid&1)
    const int eS    = tid >> 1;
    const int sBase = (tid & 1) * 8;
    float aexp[8], hstate[8];
    #pragma unroll
    for (int i = 0; i < 8; i++) {
        aexp[i]   = -__expf(A_log[eS * S_DIM + sBase + i]); // A = -exp(A_log)
        hstate[i] = 0.f;
    }
    float dtwv[R_DIM], dtbv = 0.f, dpe = 0.f;
    if (tid < E_DIM) {
        #pragma unroll
        for (int r = 0; r < R_DIM; r++) dtwv[r] = dt_w[r * E_DIM + tid];
        dtbv = dt_b[tid];
        dpe  = Dpv[tid];
    }

    for (int t = 0; t < T_DIM; t++) {
        float resid = 0.f;
        if (tid < H_DIM) {
            resid = xres[((size_t)t * N_DIM + n) * H_DIM + tid];
            xrow[tid] = resid;
        }
        __syncthreads();
        if (tid < H_DIM) {  // LayerNorm over H
            float s = 0.f, s2 = 0.f;
            for (int k = 0; k < H_DIM; k++) { float x = xrow[k]; s += x; s2 += x * x; }
            float m = s * (1.f / H_DIM), var = s2 * (1.f / H_DIM) - m * m;
            xn[tid] = (resid - m) * rsqrtf(var + 1e-5f) * mg[tid] + mb[tid];
        }
        __syncthreads();
        {   // in-projection [64 -> 256], silu on both halves
            float acc = in_b[tid];
            for (int k = 0; k < H_DIM; k++) acc += xn[k] * in_w[k * (2 * E_DIM) + tid];
            float sl = acc / (1.f + __expf(-acc));
            if (tid < E_DIM) xp[tid] = sl; else gs[tid - E_DIM] = sl;
        }
        __syncthreads();
        if (tid < R_DIM) {  // dt low-rank down-proj (+ bias per reference)
            float acc = dp_b[tid];
            for (int e = 0; e < E_DIM; e++) acc += xp[e] * dp_w[e * R_DIM + tid];
            dpr[tid] = acc;
        }
        if (tid >= 128 && tid < 128 + S_DIM) {
            int s = tid - 128;
            float acc = B_b[s];
            for (int e = 0; e < E_DIM; e++) acc += xp[e] * B_w[e * S_DIM + s];
            Bm[s] = acc;
        } else if (tid >= 128 + S_DIM && tid < 128 + 2 * S_DIM) {
            int s = tid - 128 - S_DIM;
            float acc = C_b[s];
            for (int e = 0; e < E_DIM; e++) acc += xp[e] * C_w[e * S_DIM + s];
            Cm[s] = acc;
        }
        __syncthreads();
        if (tid < E_DIM) {  // delta = softplus(dpr @ dt_w + dt_b)
            float z = dtbv;
            #pragma unroll
            for (int r = 0; r < R_DIM; r++) z += dpr[r] * dtwv[r];
            del[tid] = (z > 20.f) ? z : __logf(1.f + __expf(z));
        }
        __syncthreads();
        {   // selective scan update, 8 (e,s) cells per thread
            const float d = del[eS], xpe = xp[eS];
            float ys = 0.f;
            #pragma unroll
            for (int i = 0; i < 8; i++) {
                float ad  = __expf(d * aexp[i]);
                hstate[i] = ad * hstate[i] + d * Bm[sBase + i] * xpe;
                ys += Cm[sBase + i] * hstate[i];
            }
            ypart[tid] = ys;
        }
        __syncthreads();
        if (tid < E_DIM) {
            float y = ypart[2 * tid] + ypart[2 * tid + 1] + xp[tid] * dpe;
            yv[tid] = y * gs[tid];
        }
        __syncthreads();
        if (tid < H_DIM) {  // out projection + residual, in-place
            float acc = out_b[tid] + resid;
            for (int e = 0; e < E_DIM; e++) acc += yv[e] * out_w[e * H_DIM + tid];
            xres[((size_t)t * N_DIM + n) * H_DIM + tid] = acc;
        }
        __syncthreads();
    }
}

// ---------------------------------------------------------------------------
extern "C" void kernel_launch(void* const* d_in, const int* in_sizes, int n_in,
                              void* d_out, int out_size, void* d_ws, size_t ws_size,
                              hipStream_t stream) {
    (void)in_sizes; (void)n_in; (void)out_size; (void)ws_size;
    const float* G       = (const float*)d_in[0];
    const float* pos_emb = (const float*)d_in[1];
    const float* g1mw = (const float*)d_in[2];
    const float* g1mb = (const float*)d_in[3];
    const float* g1uw = (const float*)d_in[4];
    const float* g1ub = (const float*)d_in[5];
    const float* g1lg = (const float*)d_in[6];
    const float* g1lb = (const float*)d_in[7];
    const float* g2mw = (const float*)d_in[8];
    const float* g2mb = (const float*)d_in[9];
    const float* g2uw = (const float*)d_in[10];
    const float* g2ub = (const float*)d_in[11];
    const float* g2lg = (const float*)d_in[12];
    const float* g2lb = (const float*)d_in[13];
    const float* mlg  = (const float*)d_in[14];
    const float* mlb  = (const float*)d_in[15];
    const float* in_w = (const float*)d_in[16];
    const float* in_b = (const float*)d_in[17];
    const float* dp_w = (const float*)d_in[18];
    const float* dp_b = (const float*)d_in[19];
    const float* dt_w = (const float*)d_in[20];
    const float* dt_b = (const float*)d_in[21];
    const float* A_log= (const float*)d_in[22];
    const float* Dp   = (const float*)d_in[23];
    const float* B_w  = (const float*)d_in[24];
    const float* B_b  = (const float*)d_in[25];
    const float* C_w  = (const float*)d_in[26];
    const float* C_b  = (const float*)d_in[27];
    const float* out_w= (const float*)d_in[28];
    const float* out_b= (const float*)d_in[29];

    // workspace layout
    char* ws = (char*)d_ws;
    const size_t SZ_MSGST = (size_t)H_DIM * N_DIM * 2;                 // 256 KB
    const size_t SZ_TNH   = (size_t)T_DIM * N_DIM * H_DIM * 4;        // 16 MB
    unsigned short* msgsT = (unsigned short*)ws;
    float* agg            = (float*)(ws + SZ_MSGST);                  // agg1 then agg2
    float* x1             = (float*)(ws + SZ_MSGST + SZ_TNH);
    unsigned short* m2T   = (unsigned short*)(ws + SZ_MSGST + 2 * SZ_TNH);
    float* x2             = (float*)d_out;                            // staged in d_out

    prep_msgs_kernel<<<N_DIM, 64, 0, stream>>>(pos_emb, g1mw, g1mb, msgsT);

    gemm_graph_wmma<<<dim3(N_DIM / 128, T_DIM), 256, 0, stream>>>(G, msgsT, 0, agg);

    upd_ln_kernel<<<dim3(N_DIM, T_DIM), 64, 0, stream>>>(
        pos_emb, 0, POS_DIM, agg, H_DIM, g1uw, g1ub, g1lg, g1lb, x1);

    msgs2_kernel<<<dim3(N_DIM, T_DIM), 64, 0, stream>>>(x1, g2mw, g2mb, m2T);

    gemm_graph_wmma<<<dim3(N_DIM / 128, T_DIM), 256, 0, stream>>>(
        G, m2T, (long long)H_DIM * N_DIM, agg);

    upd_ln_kernel<<<dim3(N_DIM, T_DIM), 64, 0, stream>>>(
        x1, (long long)N_DIM * H_DIM, H_DIM, agg, H_DIM, g2uw, g2ub, g2lg, g2lb, x2);

    mamba_kernel<<<N_DIM, 256, 0, stream>>>(
        x2, mlg, mlb, in_w, in_b, dp_w, dp_b, dt_w, dt_b,
        A_log, Dp, B_w, B_b, C_w, C_b, out_w, out_b);
}
